// Block_13950053777949
// MI455X (gfx1250) — matrix operand order repaired and checked by reading, not compiled, not run on hardware
//
#include <hip/hip_runtime.h>
#include <hip/hip_bf16.h>
#include <math.h>
#include <stdint.h>

typedef __attribute__((ext_vector_type(16))) _Float16 v16h;
typedef __attribute__((ext_vector_type(8)))  float    v8f;
typedef __attribute__((ext_vector_type(4)))  unsigned int v4u;
typedef __attribute__((ext_vector_type(8)))  int v8i;
typedef __attribute__((ext_vector_type(4)))  int v4i;

constexpr int kB  = 8;
constexpr int kT  = 2048;
constexpr int kC  = 384;
constexpr int kH  = 6;
constexpr int kHS = 64;
constexpr int kBT = kB * kT;   // 16384 tokens
constexpr int kBH = kB * kH;   // 48 batch-heads
constexpr int kFF = 4 * kC;    // 1536
constexpr int kKCH = 128;      // K-chunk staged per TDM transfer

// ---------------------------------------------------------------------------
// WMMA helpers (CDNA5 16x16x32 f16 -> f32)
// ---------------------------------------------------------------------------
__device__ __forceinline__ v8f wmma_f16(v16h a, v16h b, v8f c) {
  return __builtin_amdgcn_wmma_f32_16x16x32_f16(false, a, false, b, (short)0, c,
                                                false, false);
}

// A tile 16x32 (MxK), row-major source, leading dim lda; ptr at tile origin.
__device__ __forceinline__ v16h wmma_load_a(const _Float16* __restrict__ A,
                                            int lda, int lane) {
  const int row = lane & 15;
  const int kb  = (lane < 16) ? 0 : 16;
  const _Float16* p = A + (size_t)row * lda + kb;
  v16h a;
#pragma unroll
  for (int i = 0; i < 8; ++i) a[i] = p[i];
#pragma unroll
  for (int i = 0; i < 8; ++i) a[i + 8] = p[8 + i];
  return a;
}

// B tile 32x16 (KxN) from B^T stored row-major: Bt[n][k], leading dim ldk.
__device__ __forceinline__ v16h wmma_load_bt(const _Float16* __restrict__ Bt,
                                             int ldk, int lane) {
  const int n  = lane & 15;
  const int kb = (lane < 16) ? 0 : 16;
  const _Float16* p = Bt + (size_t)n * ldk + kb;
  v16h b;
#pragma unroll
  for (int i = 0; i < 16; ++i) b[i] = p[i];
  return b;
}

__device__ __forceinline__ float gelu_f(float x) {
  const float c = 0.7978845608028654f;  // sqrt(2/pi)
  return 0.5f * x * (1.0f + tanhf(c * (x + 0.044715f * x * x * x)));
}

// ---------------------------------------------------------------------------
// 16-lane-row reductions via v_permlane16_b32 (VALU, no LDS traffic).
// Nibble i of the select = source lane (i ^ k) within each row of 16.
// ---------------------------------------------------------------------------
template <unsigned S0, unsigned S1>
__device__ __forceinline__ float permxor16(float v) {
  int i = __float_as_int(v);
  int r = __builtin_amdgcn_permlane16(i, i, S0, S1, false, false);
  return __int_as_float(r);
}

__device__ __forceinline__ float rowmax16(float v) {
  v = fmaxf(v, permxor16<0x67452301u, 0xEFCDAB89u>(v));  // xor 1
  v = fmaxf(v, permxor16<0x54761032u, 0xDCFE98BAu>(v));  // xor 2
  v = fmaxf(v, permxor16<0x32107654u, 0xBA98FEDCu>(v));  // xor 4
  v = fmaxf(v, permxor16<0xFEDCBA98u, 0x76543210u>(v));  // xor 8
  return v;
}

__device__ __forceinline__ float rowsum16(float v) {
  v += permxor16<0x67452301u, 0xEFCDAB89u>(v);
  v += permxor16<0x54761032u, 0xDCFE98BAu>(v);
  v += permxor16<0x32107654u, 0xBA98FEDCu>(v);
  v += permxor16<0xFEDCBA98u, 0x76543210u>(v);
  return v;
}

// ---------------------------------------------------------------------------
// Tensor Data Mover: stage a [nrows x kchunk] fp16 tile (row stride = kelems)
// from global into LDS. 2-D descriptor, data_size=2B, no padding/iterate.
// ---------------------------------------------------------------------------
__device__ __forceinline__ void tdm_issue_load(unsigned lds_byte,
                                               const _Float16* gsrc,
                                               int kelems, int ntotal,
                                               int kchunk, int nrows) {
  unsigned long long ga = (unsigned long long)(uintptr_t)gsrc;
  v4u g0;
  g0[0] = 1u;                                            // count=1, user desc
  g0[1] = lds_byte;                                      // lds_addr
  g0[2] = (unsigned)(ga & 0xFFFFFFFFull);                // global_addr lo
  g0[3] = (unsigned)((ga >> 32) & 0x01FFFFFFull) | (2u << 30);  // hi + type=2
  v8i g1;
  g1[0] = 1 << 16;                                       // data_size = 2 bytes
  g1[1] = (int)(((unsigned)kelems & 0xFFFFu) << 16);     // tensor_dim0 lo16
  g1[2] = (int)((((unsigned)kelems >> 16) & 0xFFFFu) |
                (((unsigned)ntotal & 0xFFFFu) << 16));   // dim0 hi | dim1 lo
  g1[3] = (int)((((unsigned)ntotal >> 16) & 0xFFFFu) |
                (((unsigned)kchunk & 0xFFFFu) << 16));   // dim1 hi | tile_dim0
  g1[4] = (int)((unsigned)nrows & 0xFFFFu);              // tile_dim1
  g1[5] = kelems;                                        // dim0_stride lo32
  g1[6] = 0;
  g1[7] = 0;
  v4i gz = {0, 0, 0, 0};
#if __clang_major__ >= 23
  v8i gz8 = {0, 0, 0, 0, 0, 0, 0, 0};
  __builtin_amdgcn_tensor_load_to_lds(g0, g1, gz, gz, gz8, 0);
#else
  __builtin_amdgcn_tensor_load_to_lds(g0, g1, gz, gz, 0);
#endif
}

// ---------------------------------------------------------------------------
// Weight conversion kernels (fp32 -> fp16, transposed for contiguous B loads)
// ---------------------------------------------------------------------------
__global__ void cvt_qkv_t(const float* __restrict__ W, _Float16* __restrict__ o) {
  int i = blockIdx.x * blockDim.x + threadIdx.x;
  if (i >= kC * kC) return;
  int n = i / kC, k = i % kC;
  int h = n >> 6, d = n & (kHS - 1);
  o[i] = (_Float16)W[((size_t)(h * kC + k)) * kHS + d];
}

__global__ void cvt_t(const float* __restrict__ W, _Float16* __restrict__ o,
                      int K, int N) {
  int i = blockIdx.x * blockDim.x + threadIdx.x;
  if (i >= K * N) return;
  int n = i / K, k = i % K;
  o[i] = (_Float16)W[(size_t)k * N + n];
}

// ---------------------------------------------------------------------------
// LayerNorm (bias-free, biased variance, eps=1e-5) -> fp16
// ---------------------------------------------------------------------------
__global__ __launch_bounds__(128) void layernorm_kernel(
    const float* __restrict__ x, const float* __restrict__ w,
    _Float16* __restrict__ out) {
  __shared__ float ssum[128], ssq[128];
  const int row = blockIdx.x;
  const float* xr = x + (size_t)row * kC;
  float s = 0.f, q = 0.f;
  for (int i = threadIdx.x; i < kC; i += 128) {
    float v = xr[i];
    s += v;
    q += v * v;
  }
  ssum[threadIdx.x] = s;
  ssq[threadIdx.x] = q;
  __syncthreads();
  for (int off = 64; off > 0; off >>= 1) {
    if (threadIdx.x < off) {
      ssum[threadIdx.x] += ssum[threadIdx.x + off];
      ssq[threadIdx.x] += ssq[threadIdx.x + off];
    }
    __syncthreads();
  }
  const float mu = ssum[0] * (1.0f / kC);
  const float var = ssq[0] * (1.0f / kC) - mu * mu;
  const float inv = rsqrtf(var + 1e-5f);
  for (int i = threadIdx.x; i < kC; i += 128)
    out[(size_t)row * kC + i] = (_Float16)((xr[i] - mu) * inv * w[i]);
}

// ---------------------------------------------------------------------------
// WMMA GEMM with TDM-staged weights: block = 8 waves -> 128x64 output panel.
// B strip (64 cols x 128-k chunk, 16KB) is DMA'd into LDS by wave 0 via
// tensor_load_to_lds, double-buffered, and shared by all 8 waves.
// Epilogues: 0 = Q/K permute -> [BH,T,hs] fp16
//            1 = V permute   -> [BH,hs,T] fp16
//            2 = + residual + bias -> f32
//            3 = gelu -> fp16
//            4 = + residual -> f32
// ---------------------------------------------------------------------------
template <int EPI>
__global__ __launch_bounds__(256) void gemm_wmma_kernel(
    const _Float16* __restrict__ A, const _Float16* __restrict__ Bt,
    const float* __restrict__ res, const float* __restrict__ bias,
    float* __restrict__ outf, _Float16* __restrict__ outh, int M, int N, int K) {
  __shared__ _Float16 Bls[2][64 * kKCH];  // 2 x 16KB double buffer

  const int lane = threadIdx.x & 31;
  const int wv = threadIdx.x >> 5;
  const int nstrips = N >> 6;
  const int mblk = blockIdx.x / nstrips;
  const int nstrip = blockIdx.x % nstrips;
  const int m0 = mblk * 128 + wv * 16;
  const int n0 = nstrip * 64;
  const int nch = K / kKCH;

  if (threadIdx.x < 32) {
    tdm_issue_load((unsigned)(uintptr_t)&Bls[0][0], Bt + (size_t)n0 * K,
                   K, N, kKCH, 64);
  }

  v8f acc[4] = {};
  for (int c = 0; c < nch; ++c) {
    if (threadIdx.x < 32) __builtin_amdgcn_s_wait_tensorcnt(0);
    __syncthreads();  // publish chunk c; all waves done with buf[(c+1)&1]
    if (threadIdx.x < 32 && c + 1 < nch) {
      tdm_issue_load((unsigned)(uintptr_t)&Bls[(c + 1) & 1][0],
                     Bt + (size_t)n0 * K + (size_t)(c + 1) * kKCH,
                     K, N, kKCH, 64);
    }
    const _Float16* bbuf = &Bls[c & 1][0];
#pragma unroll
    for (int ks = 0; ks < kKCH; ks += 32) {
      const int k0 = c * kKCH + ks;
      v16h a = wmma_load_a(A + (size_t)m0 * K + k0, K, lane);
      const int bcol = lane & 15;
      const int bkb = (lane < 16) ? 0 : 16;
#pragma unroll
      for (int j = 0; j < 4; ++j) {
        const _Float16* bp = bbuf + (size_t)(16 * j + bcol) * kKCH + ks + bkb;
        v16h b;
#pragma unroll
        for (int i = 0; i < 16; ++i) b[i] = bp[i];
        acc[j] = wmma_f16(a, b, acc[j]);
      }
    }
  }

  const int col = lane & 15;
  const int rb = (lane < 16) ? 0 : 8;
#pragma unroll
  for (int j = 0; j < 4; ++j) {
    const int n = n0 + 16 * j + col;
#pragma unroll
    for (int e = 0; e < 8; ++e) {
      const int m = m0 + rb + e;
      const float v = acc[j][e];
      const size_t idx = (size_t)m * N + n;
      if (EPI == 0) {
        const int b = m >> 11, t = m & (kT - 1);
        const int h = n >> 6, d = n & (kHS - 1);
        outh[((size_t)(b * kH + h) * kT + t) * kHS + d] = (_Float16)v;
      } else if (EPI == 1) {
        const int b = m >> 11, t = m & (kT - 1);
        const int h = n >> 6, d = n & (kHS - 1);
        outh[((size_t)(b * kH + h) * kHS + d) * kT + t] = (_Float16)v;
      } else if (EPI == 2) {
        outf[idx] = v + res[idx] + bias[n];
      } else if (EPI == 3) {
        outh[idx] = (_Float16)gelu_f(v);
      } else {
        outf[idx] = v + res[idx];
      }
    }
  }
}

// ---------------------------------------------------------------------------
// Flash attention (non-causal, scale = C^-0.5). One wave = 16 query rows,
// streaming keys in 32-wide tiles. Online softmax with permlane16 row-max;
// row-sum reduction deferred to the end (alpha is row-uniform, so per-lane
// partial l is exact).
// ---------------------------------------------------------------------------
__global__ __launch_bounds__(128) void flash_attn_kernel(
    const _Float16* __restrict__ Q,   // [BH, T, hs]
    const _Float16* __restrict__ Kk,  // [BH, T, hs]
    const _Float16* __restrict__ Vt,  // [BH, hs, T]
    _Float16* __restrict__ O)         // [B, T, C], head h at cols h*64..
{
  const int lane = threadIdx.x & 31;
  const int wv = threadIdx.x >> 5;
  const int bh = blockIdx.y;
  const int q0 = (blockIdx.x * 4 + wv) * 16;
  const _Float16* Qb = Q + (size_t)bh * kT * kHS;
  const _Float16* Kb = Kk + (size_t)bh * kT * kHS;
  const _Float16* Vb = Vt + (size_t)bh * kHS * kT;

  __shared__ _Float16 Pb[4][16 * 32];
  _Float16* myP = &Pb[wv][0];

  const v16h qa0 = wmma_load_a(Qb + (size_t)q0 * kHS + 0, kHS, lane);
  const v16h qa1 = wmma_load_a(Qb + (size_t)q0 * kHS + 32, kHS, lane);

  v8f o0 = {}, o1 = {}, o2 = {}, o3 = {};
  float mrow[8], lacc[8];
#pragma unroll
  for (int e = 0; e < 8; ++e) { mrow[e] = -3.0e38f; lacc[e] = 0.f; }

  const float scale = 0.05103103630798288f;  // 384^-0.5
  const int col = lane & 15;
  const int rb = (lane < 16) ? 0 : 8;

  for (int s0 = 0; s0 < kT; s0 += 32) {
    v8f st0 = {}, st1 = {};
    {
      v16h b00 = wmma_load_bt(Kb + (size_t)(s0 + 0) * kHS + 0, kHS, lane);
      v16h b01 = wmma_load_bt(Kb + (size_t)(s0 + 0) * kHS + 32, kHS, lane);
      v16h b10 = wmma_load_bt(Kb + (size_t)(s0 + 16) * kHS + 0, kHS, lane);
      v16h b11 = wmma_load_bt(Kb + (size_t)(s0 + 16) * kHS + 32, kHS, lane);
      st0 = wmma_f16(qa0, b00, st0);
      st0 = wmma_f16(qa1, b01, st0);
      st1 = wmma_f16(qa0, b10, st1);
      st1 = wmma_f16(qa1, b11, st1);
    }
    float alpha[8];
#pragma unroll
    for (int e = 0; e < 8; ++e) {
      const float a0 = st0[e] * scale;
      const float a1 = st1[e] * scale;
      const float cur = rowmax16(fmaxf(a0, a1));   // VALU permlane reduce
      const float mn = fmaxf(mrow[e], cur);
      alpha[e] = __expf(mrow[e] - mn);
      const float p0 = __expf(a0 - mn);
      const float p1 = __expf(a1 - mn);
      lacc[e] = lacc[e] * alpha[e] + (p0 + p1);    // per-lane partial row sum
      mrow[e] = mn;
      myP[(rb + e) * 32 + col] = (_Float16)p0;
      myP[(rb + e) * 32 + 16 + col] = (_Float16)p1;
    }
#pragma unroll
    for (int e = 0; e < 8; ++e) {
      o0[e] *= alpha[e];
      o1[e] *= alpha[e];
      o2[e] *= alpha[e];
      o3[e] *= alpha[e];
    }
    // wave-private LDS round-trip: C-layout -> A-layout for P
    asm volatile("s_wait_dscnt 0" ::: "memory");
    v16h pa;
    {
      const int arow = lane & 15;
      const int akb = (lane < 16) ? 0 : 16;
      const _Float16* lp = myP + arow * 32 + akb;
#pragma unroll
      for (int i = 0; i < 8; ++i) pa[i] = lp[i];
#pragma unroll
      for (int i = 0; i < 8; ++i) pa[8 + i] = lp[8 + i];
    }
    v16h vb0 = wmma_load_bt(Vb + (size_t)0 * kT + s0, kT, lane);
    v16h vb1 = wmma_load_bt(Vb + (size_t)16 * kT + s0, kT, lane);
    v16h vb2 = wmma_load_bt(Vb + (size_t)32 * kT + s0, kT, lane);
    v16h vb3 = wmma_load_bt(Vb + (size_t)48 * kT + s0, kT, lane);
    o0 = wmma_f16(pa, vb0, o0);
    o1 = wmma_f16(pa, vb1, o1);
    o2 = wmma_f16(pa, vb2, o2);
    o3 = wmma_f16(pa, vb3, o3);
  }

  const int b = bh / kH, h = bh % kH;
#pragma unroll
  for (int e = 0; e < 8; ++e) {
    const float lr = rowsum16(lacc[e]);
    const float inv = 1.0f / lr;
    const int t = q0 + rb + e;
    const size_t base = ((size_t)(b * kT + t)) * kC + h * kHS;
    O[base + 0 + col] = (_Float16)(o0[e] * inv);
    O[base + 16 + col] = (_Float16)(o1[e] * inv);
    O[base + 32 + col] = (_Float16)(o2[e] * inv);
    O[base + 48 + col] = (_Float16)(o3[e] * inv);
  }
}

// ---------------------------------------------------------------------------
extern "C" void kernel_launch(void* const* d_in, const int* in_sizes, int n_in,
                              void* d_out, int out_size, void* d_ws,
                              size_t ws_size, hipStream_t stream) {
  (void)in_sizes; (void)n_in; (void)out_size; (void)ws_size;
  const float* x    = (const float*)d_in[0];
  const float* ln1w = (const float*)d_in[1];
  const float* ln2w = (const float*)d_in[2];
  const float* Wq   = (const float*)d_in[3];
  const float* Wk   = (const float*)d_in[4];
  const float* Wv   = (const float*)d_in[5];
  const float* Wo   = (const float*)d_in[6];
  const float* bo   = (const float*)d_in[7];
  const float* W1   = (const float*)d_in[8];
  const float* W2   = (const float*)d_in[9];
  float* out = (float*)d_out;

  char* ws = (char*)d_ws;
  size_t off = 0;
  auto take = [&](size_t bytes) -> char* {
    char* p = ws + off;
    off = (off + bytes + 255) & ~(size_t)255;
    return p;
  };
  _Float16* h16  = (_Float16*)take((size_t)kBT * kC * 2);
  _Float16* q16  = (_Float16*)take((size_t)kBH * kT * kHS * 2);
  _Float16* k16  = (_Float16*)take((size_t)kBH * kT * kHS * 2);
  _Float16* vt16 = (_Float16*)take((size_t)kBH * kT * kHS * 2);
  _Float16* ao16 = (_Float16*)take((size_t)kBT * kC * 2);
  float*    xo   = (float*)   take((size_t)kBT * kC * 4);
  _Float16* h2   = (_Float16*)take((size_t)kBT * kC * 2);
  _Float16* a1   = (_Float16*)take((size_t)kBT * kFF * 2);
  _Float16* wqT  = (_Float16*)take((size_t)kC * kC * 2);
  _Float16* wkT  = (_Float16*)take((size_t)kC * kC * 2);
  _Float16* wvT  = (_Float16*)take((size_t)kC * kC * 2);
  _Float16* woT  = (_Float16*)take((size_t)kC * kC * 2);
  _Float16* w1T  = (_Float16*)take((size_t)kC * kFF * 2);
  _Float16* w2T  = (_Float16*)take((size_t)kFF * kC * 2);

  {
    int n = kC * kC, gr = (n + 255) / 256;
    cvt_qkv_t<<<gr, 256, 0, stream>>>(Wq, wqT);
    cvt_qkv_t<<<gr, 256, 0, stream>>>(Wk, wkT);
    cvt_qkv_t<<<gr, 256, 0, stream>>>(Wv, wvT);
    cvt_t<<<gr, 256, 0, stream>>>(Wo, woT, kC, kC);
  }
  {
    int n = kC * kFF, gr = (n + 255) / 256;
    cvt_t<<<gr, 256, 0, stream>>>(W1, w1T, kC, kFF);
    cvt_t<<<gr, 256, 0, stream>>>(W2, w2T, kFF, kC);
  }

  auto gemm_blocks = [](int M, int N) { return (M / 128) * (N / 64); };

  layernorm_kernel<<<kBT, 128, 0, stream>>>(x, ln1w, h16);
  gemm_wmma_kernel<0><<<gemm_blocks(kBT, kC), 256, 0, stream>>>(
      h16, wqT, nullptr, nullptr, nullptr, q16, kBT, kC, kC);
  gemm_wmma_kernel<0><<<gemm_blocks(kBT, kC), 256, 0, stream>>>(
      h16, wkT, nullptr, nullptr, nullptr, k16, kBT, kC, kC);
  gemm_wmma_kernel<1><<<gemm_blocks(kBT, kC), 256, 0, stream>>>(
      h16, wvT, nullptr, nullptr, nullptr, vt16, kBT, kC, kC);
  dim3 ag(kT / 64, kBH);
  flash_attn_kernel<<<ag, 128, 0, stream>>>(q16, k16, vt16, ao16);
  gemm_wmma_kernel<2><<<gemm_blocks(kBT, kC), 256, 0, stream>>>(
      ao16, woT, x, bo, xo, nullptr, kBT, kC, kC);
  layernorm_kernel<<<kBT, 128, 0, stream>>>(xo, ln2w, h2);
  gemm_wmma_kernel<3><<<gemm_blocks(kBT, kFF), 256, 0, stream>>>(
      h2, w1T, nullptr, nullptr, nullptr, a1, kBT, kFF, kC);
  gemm_wmma_kernel<4><<<gemm_blocks(kBT, kC), 256, 0, stream>>>(
      a1, w2T, xo, nullptr, out, nullptr, kBT, kC, kFF);
}
